// GCN_BiLSTMK_91156385890780
// MI455X (gfx1250) — compile-verified, matrix operand order checked by reading, not run on hardware
//
#include <hip/hip_runtime.h>
#include <math.h>

#define NN  20000
#define EE  100000
#define FN  26
#define FE  21
#define H1V 64
#define H2V 30

typedef __attribute__((ext_vector_type(16))) _Float16 v16h;
typedef __attribute__((ext_vector_type(2)))  _Float16 h2v;
typedef __attribute__((ext_vector_type(8)))  float    v8f;

__device__ __forceinline__ float sigmoidf_(float x){ return 1.f/(1.f+__expf(-x)); }
__device__ __forceinline__ float leakyf_(float x){ return x > 0.f ? x : 0.01f*x; }

// ---------------- utility: zero ----------------
__global__ void k_zero(float* __restrict__ p, int n){
  int i = blockIdx.x*blockDim.x + threadIdx.x;
  if (i < n) p[i] = 0.f;
}

// ---------------- in-degree counts ----------------
__global__ void k_count(const int* __restrict__ dst, float* __restrict__ cnt){
  int e = blockIdx.x*blockDim.x + threadIdx.x;
  if (e < EE) atomicAdd(&cnt[dst[e]], 1.0f);
}

// ---------------- pack combined edge weight into WMMA B-fragment order ----------------
// K ordering: K = i*32 + k (k padded 22 -> 32 with zeros; division-free on the hot path).
// Wc[K,n] = ew[(i*FOUT+n)*FE + k] (k<FE) ; eb[i*FOUT+n] (k==FE) ; 0 (k>FE or n>=FOUT).
__device__ __forceinline__ float wcval(const float* __restrict__ ew, const float* __restrict__ eb,
                                       int K, int n, int FIN, int FOUT){
  int i = K >> 5;
  int k = K & 31;
  if (n >= FOUT || i >= FIN || k > FE) return 0.f;
  int row = i*FOUT + n;
  return (k < FE) ? ew[row*FE + k] : eb[row];
}

__global__ void k_prep_wc(const float* __restrict__ ew, const float* __restrict__ eb,
                          unsigned* __restrict__ wc, int FIN, int FOUT,
                          int NTILES, int KCHUNKS){
  int total = KCHUNKS*NTILES*256;
  int idx = blockIdx.x*blockDim.x + threadIdx.x;
  if (idx >= total) return;
  int j    = idx & 7;
  int lane = (idx >> 3) & 31;
  int ct   = idx >> 8;
  int t = ct % NTILES;
  int c = ct / NTILES;
  int hi = lane >> 4, nl = lane & 15;
  int n  = t*16 + nl;
  int K0 = c*32 + (j < 4 ? 8*hi + 2*j : 16 + 8*hi + 2*(j-4));
  union { _Float16 h[2]; unsigned u; } P;
  P.h[0] = (_Float16)wcval(ew, eb, K0,   n, FIN, FOUT);
  P.h[1] = (_Float16)wcval(ew, eb, K0+1, n, FIN, FOUT);
  wc[idx] = P.u;
}

// ---------------- fused NNConv edge messages via WMMA ----------------
// msg[E, FOUT] = (x[src] (x) [ea,1,pad]) [E, FIN*32] @ Wc [FIN*32, FOUT]; scatter-add by dst.
// K-chunk c == node-feature i: A fragment = splat(x[c]) * fixed per-lane edge-feature pairs.
// E is a multiple of 16, so a 16-edge tile is entirely valid or entirely invalid (uniform).
template<int FIN, int FOUT, int NTILES>
__global__ void __launch_bounds__(128)
k_edge_msg(const float* __restrict__ xin, const int* __restrict__ src,
           const int* __restrict__ dst, const float* __restrict__ ea,
           const unsigned* __restrict__ wc, float* __restrict__ acc){
  const int FOUTP = NTILES*16;
  int wave  = threadIdx.x >> 5;
  int lane  = threadIdx.x & 31;
  int tile  = blockIdx.x*4 + wave;
  int ebase = tile*16;
  bool tileValid = (ebase < EE);   // wave-uniform

  __shared__ _Float16 xs[4][16][FIN];   // f16 node features of gathered sources
  __shared__ _Float16 eas[4][16][32];   // f16 augmented edge features, zero-padded to 32
  __shared__ int      dsh[4][16];

  if (tileValid){
    for (int idx = lane; idx < 16*FIN; idx += 32){
      int m = idx / FIN, f = idx - m*FIN;
      xs[wave][m][f] = (_Float16)xin[src[ebase + m]*FIN + f];
    }
    for (int idx = lane; idx < 16*32; idx += 32){
      int m = idx >> 5, k = idx & 31;
      float v = (k < FE) ? ea[(ebase + m)*FE + k] : (k == FE ? 1.0f : 0.f);
      eas[wave][m][k] = (_Float16)v;
    }
    if (lane < 16) dsh[wave][lane] = dst[ebase + lane];
  } else {
    for (int idx = lane; idx < 16*FIN; idx += 32){
      int m = idx / FIN, f = idx - m*FIN;
      xs[wave][m][f] = (_Float16)0.f;
    }
    for (int idx = lane; idx < 16*32; idx += 32){
      int m = idx >> 5, k = idx & 31;
      eas[wave][m][k] = (_Float16)0.f;
    }
    if (lane < 16) dsh[wave][lane] = 0;
  }
  __syncthreads();

  int m  = lane & 15;      // A-fragment row (edge within tile)
  int hi = lane >> 4;
  const _Float16* xrow = xs[wave][m];
  const _Float16* erow = eas[wave][m];

  // This lane's fixed packed edge-feature pairs (even k offsets -> aligned dwords).
  h2v ae[8];
  #pragma unroll
  for (int j = 0; j < 8; ++j){
    int k0 = (j < 4) ? 8*hi + 2*j : 16 + 8*hi + 2*(j-4);
    ae[j] = *(const h2v*)&erow[k0];
  }

  v8f cacc[NTILES];
  #pragma unroll
  for (int t = 0; t < NTILES; ++t){ v8f z = {0,0,0,0,0,0,0,0}; cacc[t] = z; }

  for (int c = 0; c < FIN; ++c){
    _Float16 xh = xrow[c];
    h2v xsplat = {xh, xh};
    union { v16h v; h2v p[8]; } A;
    #pragma unroll
    for (int j = 0; j < 8; ++j) A.p[j] = xsplat * ae[j];   // v_pk_mul_f16
    #pragma unroll
    for (int t = 0; t < NTILES; ++t){
      union { v16h v; unsigned u[8]; } B;
      const unsigned* bp = wc + ((size_t)(c*NTILES + t)*32 + lane)*8;
      #pragma unroll
      for (int j = 0; j < 8; ++j) B.u[j] = bp[j];
      cacc[t] = __builtin_amdgcn_wmma_f32_16x16x32_f16(false, A.v, false, B.v,
                                                       (short)0, cacc[t], false, false);
    }
  }

  // Scatter-add. C/D layout: lane holds column n = lane&15, rows mrow = 8*hi + r.
  if (tileValid){
    int drow[8];
    #pragma unroll
    for (int r = 0; r < 8; ++r) drow[r] = dsh[wave][8*hi + r];
    int ncol = lane & 15;
    #pragma unroll
    for (int t = 0; t < NTILES; ++t){
      int n = t*16 + ncol;
      if (n < FOUT){   // compile-time true except layer-2 tile 1 (cols >= 30)
        #pragma unroll
        for (int r = 0; r < 8; ++r)
          atomicAdd(&acc[(size_t)drow[r]*FOUTP + n], cacc[t][r]);
      }
    }
  }
}

// ---------------- node update 1: mean-aggr + root GEMM + LN + leaky ----------------
__global__ void k_node1(const float* __restrict__ x, const float* __restrict__ acc,
                        const float* __restrict__ cnt, const float* __restrict__ root,
                        const float* __restrict__ bias, const float* __restrict__ g,
                        const float* __restrict__ be, float* __restrict__ hout){
  int n = blockIdx.x, o = threadIdx.x; // blockDim = 64
  float v = acc[(size_t)n*64 + o] / fmaxf(cnt[n], 1.f);
  float dot = 0.f;
  #pragma unroll
  for (int i = 0; i < FN; ++i) dot += x[n*FN + i]*root[i*H1V + o];
  v += dot + bias[o];
  __shared__ float red[64];
  red[o] = v; __syncthreads();
  for (int s = 32; s > 0; s >>= 1){ if (o < s) red[o] += red[o+s]; __syncthreads(); }
  float mean = red[0]*(1.f/64.f);
  __syncthreads();
  float d = v - mean;
  red[o] = d*d; __syncthreads();
  for (int s = 32; s > 0; s >>= 1){ if (o < s) red[o] += red[o+s]; __syncthreads(); }
  float var = red[0]*(1.f/64.f);
  float y = d*rsqrtf(var + 1e-5f)*g[o] + be[o];
  hout[(size_t)n*64 + o] = leakyf_(y);
}

// ---------------- node update 2 (30 valid of 32 lanes; output padded to stride 32) ----------------
__global__ void k_node2(const float* __restrict__ h1, const float* __restrict__ acc,
                        const float* __restrict__ cnt, const float* __restrict__ root,
                        const float* __restrict__ bias, const float* __restrict__ g,
                        const float* __restrict__ be, float* __restrict__ hout){
  int n = blockIdx.x, o = threadIdx.x; // blockDim = 32
  bool valid = o < H2V;
  float v = 0.f;
  if (valid){
    v = acc[(size_t)n*32 + o] / fmaxf(cnt[n], 1.f);
    float dot = 0.f;
    #pragma unroll
    for (int i = 0; i < H1V; ++i) dot += h1[(size_t)n*H1V + i]*root[i*H2V + o];
    v += dot + bias[o];
  }
  __shared__ float red[32];
  red[o] = valid ? v : 0.f; __syncthreads();
  for (int s = 16; s > 0; s >>= 1){ if (o < s) red[o] += red[o+s]; __syncthreads(); }
  float mean = red[0]*(1.f/30.f);
  __syncthreads();
  float d = v - mean;
  red[o] = valid ? d*d : 0.f; __syncthreads();
  for (int s = 16; s > 0; s >>= 1){ if (o < s) red[o] += red[o+s]; __syncthreads(); }
  float var = red[0]*(1.f/30.f);
  int oc = valid ? o : 0;
  float y = d*rsqrtf(var + 1e-5f)*g[oc] + be[oc];
  hout[(size_t)n*32 + o] = valid ? leakyf_(y) : 0.f;
}

// ---------------- BiLSTM: 2 blocks (fwd/bwd), 256 threads = 256 gate rows ----------------
__global__ void __launch_bounds__(256)
k_bilstm(const float* __restrict__ h2,
         const float* __restrict__ WihF, const float* __restrict__ WhhF,
         const float* __restrict__ bihF, const float* __restrict__ bhhF,
         const float* __restrict__ WihB, const float* __restrict__ WhhB,
         const float* __restrict__ bihB, const float* __restrict__ bhhB,
         float* __restrict__ hf, float* __restrict__ hb){
  int dir = blockIdx.x;
  const float* Wih = dir ? WihB : WihF;
  const float* Whh = dir ? WhhB : WhhF;
  const float* bih = dir ? bihB : bihF;
  const float* bhh = dir ? bhhB : bhhF;
  float* out = dir ? hb : hf;
  int gI = threadIdx.x;
  float wih[H2V];
  #pragma unroll
  for (int j = 0; j < H2V; ++j) wih[j] = Wih[gI*H2V + j];
  float whh[H1V];
  #pragma unroll
  for (int j = 0; j < H1V; ++j) whh[j] = Whh[gI*H1V + j];
  float bsum = bih[gI] + bhh[gI];
  __shared__ float hsh[H1V], csh[H1V], xsh[H2V], gsh[256];
  if (gI < H1V){ hsh[gI] = 0.f; csh[gI] = 0.f; }
  __syncthreads();
  for (int s = 0; s < NN; ++s){
    int t = dir ? (NN - 1 - s) : s;
    if (gI < H2V) xsh[gI] = h2[(size_t)t*32 + gI];
    __syncthreads();
    float a = bsum;
    #pragma unroll
    for (int j = 0; j < H2V; ++j) a += xsh[j]*wih[j];
    #pragma unroll
    for (int j = 0; j < H1V; ++j) a += hsh[j]*whh[j];
    gsh[gI] = a;
    __syncthreads();
    if (gI < H1V){
      float ig = sigmoidf_(gsh[gI]);
      float fg = sigmoidf_(gsh[64 + gI]);
      float gg = tanhf(gsh[128 + gI]);
      float og = sigmoidf_(gsh[192 + gI]);
      float c = fg*csh[gI] + ig*gg;
      csh[gI] = c;
      float h = og*tanhf(c);
      hsh[gI] = h;
      out[(size_t)t*H1V + gI] = h;
    }
    __syncthreads();
  }
}

// ---------------- final: LN(128) + FC(128->2) ----------------
__global__ void k_final(const float* __restrict__ hf, const float* __restrict__ hb,
                        const float* __restrict__ gl, const float* __restrict__ bl,
                        const float* __restrict__ fcw, const float* __restrict__ fcb,
                        float* __restrict__ out){
  int n = blockIdx.x, o = threadIdx.x; // blockDim = 128
  float v = (o < 64) ? hf[(size_t)n*64 + o] : hb[(size_t)n*64 + (o - 64)];
  __shared__ float red[128];
  red[o] = v; __syncthreads();
  for (int s = 64; s > 0; s >>= 1){ if (o < s) red[o] += red[o+s]; __syncthreads(); }
  float mean = red[0]*(1.f/128.f);
  __syncthreads();
  float d = v - mean;
  red[o] = d*d; __syncthreads();
  for (int s = 64; s > 0; s >>= 1){ if (o < s) red[o] += red[o+s]; __syncthreads(); }
  float var = red[0]*(1.f/128.f);
  float y = d*rsqrtf(var + 1e-5f)*gl[o] + bl[o];
  __syncthreads();
  red[o] = y*fcw[o]; __syncthreads();
  for (int s = 64; s > 0; s >>= 1){ if (o < s) red[o] += red[o+s]; __syncthreads(); }
  if (o == 0) out[n*2 + 0] = red[0] + fcb[0];
  __syncthreads();
  red[o] = y*fcw[128 + o]; __syncthreads();
  for (int s = 64; s > 0; s >>= 1){ if (o < s) red[o] += red[o+s]; __syncthreads(); }
  if (o == 0) out[n*2 + 1] = red[0] + fcb[1];
}

extern "C" void kernel_launch(void* const* d_in, const int* in_sizes, int n_in,
                              void* d_out, int out_size, void* d_ws, size_t ws_size,
                              hipStream_t stream){
  (void)in_sizes; (void)n_in; (void)out_size; (void)ws_size;
  const float* x    = (const float*)d_in[0];
  const int*   ei   = (const int*)  d_in[1];
  const float* ea   = (const float*)d_in[2];
  const float* ew1  = (const float*)d_in[3];
  const float* eb1  = (const float*)d_in[4];
  const float* root1= (const float*)d_in[5];
  const float* b1   = (const float*)d_in[6];
  const float* g1   = (const float*)d_in[7];
  const float* be1  = (const float*)d_in[8];
  const float* ew2  = (const float*)d_in[9];
  const float* eb2  = (const float*)d_in[10];
  const float* root2= (const float*)d_in[11];
  const float* b2   = (const float*)d_in[12];
  const float* g2   = (const float*)d_in[13];
  const float* be2  = (const float*)d_in[14];
  const float* WihF = (const float*)d_in[15];
  const float* WhhF = (const float*)d_in[16];
  const float* bihF = (const float*)d_in[17];
  const float* bhhF = (const float*)d_in[18];
  const float* WihB = (const float*)d_in[19];
  const float* WhhB = (const float*)d_in[20];
  const float* bihB = (const float*)d_in[21];
  const float* bhhB = (const float*)d_in[22];
  const float* gl   = (const float*)d_in[23];
  const float* bl   = (const float*)d_in[24];
  const float* fcw  = (const float*)d_in[25];
  const float* fcb  = (const float*)d_in[26];
  const int* srcI = ei;
  const int* dstI = ei + EE;
  float* out = (float*)d_out;

  char* p = (char*)d_ws;
  auto alloc = [&](size_t bytes)->void* {
    void* r = (void*)p;
    p += (bytes + 255) & ~(size_t)255;
    return r;
  };
  const int KC1 = FN,  NT1 = 4;   // K = 26*32 = 832  -> 26 chunks; N = 64 -> 4 tiles
  const int KC2 = H1V, NT2 = 2;   // K = 64*32 = 2048 -> 64 chunks; N = 30 -> 2 tiles (pad 32)
  unsigned* wc1 = (unsigned*)alloc((size_t)KC1*NT1*256*4);
  unsigned* wc2 = (unsigned*)alloc((size_t)KC2*NT2*256*4);
  float* cnt  = (float*)alloc((size_t)NN*4);
  float* acc1 = (float*)alloc((size_t)NN*64*4);
  float* acc2 = (float*)alloc((size_t)NN*32*4);
  float* h1   = (float*)alloc((size_t)NN*64*4);
  float* h2   = (float*)alloc((size_t)NN*32*4);
  float* hfv  = (float*)alloc((size_t)NN*64*4);
  float* hbv  = (float*)alloc((size_t)NN*64*4);

  // zero atomic accumulators (harness does not re-zero between replays)
  k_zero<<<(NN      + 255)/256, 256, 0, stream>>>(cnt,  NN);
  k_zero<<<(NN*64   + 255)/256, 256, 0, stream>>>(acc1, NN*64);
  k_zero<<<(NN*32   + 255)/256, 256, 0, stream>>>(acc2, NN*32);

  // pre-swizzle combined edge-weight matrices into WMMA B-fragment order (f16)
  k_prep_wc<<<(KC1*NT1*256 + 255)/256, 256, 0, stream>>>(ew1, eb1, wc1, FN,  H1V, NT1, KC1);
  k_prep_wc<<<(KC2*NT2*256 + 255)/256, 256, 0, stream>>>(ew2, eb2, wc2, H1V, H2V, NT2, KC2);

  // in-degree counts
  k_count<<<(EE + 255)/256, 256, 0, stream>>>(dstI, cnt);

  int tiles  = (EE + 15)/16;          // 6250 16-edge tiles
  int blocks = (tiles + 3)/4;         // 4 waves per 128-thread block

  // layer 1: fused bilinear edge GEMM + scatter-add, then node update
  k_edge_msg<FN, H1V, NT1><<<blocks, 128, 0, stream>>>(x, srcI, dstI, ea, wc1, acc1);
  k_node1<<<NN, 64, 0, stream>>>(x, acc1, cnt, root1, b1, g1, be1, h1);

  // layer 2
  k_edge_msg<H1V, H2V, NT2><<<blocks, 128, 0, stream>>>(h1, srcI, dstI, ea, wc2, acc2);
  k_node2<<<NN, 32, 0, stream>>>(h1, acc2, cnt, root2, b2, g2, be2, h2);

  // BiLSTM (sequential recurrence; fwd+bwd concurrent)
  k_bilstm<<<2, 256, 0, stream>>>(h2, WihF, WhhF, bihF, bhhF, WihB, WhhB, bihB, bhhB, hfv, hbv);

  // final LayerNorm + FC
  k_final<<<NN, 128, 0, stream>>>(hfv, hbv, gl, bl, fcw, fcb, out);
}
